// MultiHeadAttention_41875931136447
// MI455X (gfx1250) — matrix operand order repaired and checked by reading, not compiled, not run on hardware
//
#include <hip/hip_runtime.h>
#include <hip/hip_bf16.h>

// MI455X / gfx1250, wave32. All matmuls via v_wmma_f32_16x16x32_f16.

#define NH 8
#define S_LEN 2048

typedef _Float16 v16h __attribute__((ext_vector_type(16)));
typedef _Float16 v8h  __attribute__((ext_vector_type(8)));
typedef float    v8f  __attribute__((ext_vector_type(8)));
typedef float    v4f  __attribute__((ext_vector_type(4)));
typedef int      v4i  __attribute__((ext_vector_type(4)));

__device__ __forceinline__ v8f wmma16(v16h a, v16h b, v8f c) {
    // D = A(16x32 f16) * B(32x16 f16) + C(16x16 f32)
    return __builtin_amdgcn_wmma_f32_16x16x32_f16(false, a, false, b,
                                                  (short)0, c, false, false);
}

// ---------------------------------------------------------------------------
// Kernel 1: pack weights into WMMA B-fragment lane order (f16).
// B-frag 32x16: lane L -> col = L&15; halves h -> K = (L>>4)*8 + (h<8 ? h : 8+h).
// QKV weights: [256][128], 8 k-chunks x 8 col-tiles.
// Wfc: [128][256], 4 k-chunks x 16 col-tiles.
// ---------------------------------------------------------------------------
__global__ __launch_bounds__(256) void pack_weights_kernel(
    const float* __restrict__ Wq, const float* __restrict__ Wk,
    const float* __restrict__ Wv, const float* __restrict__ Wfc,
    _Float16* __restrict__ wsW) {
    int idx = blockIdx.x * 256 + threadIdx.x;   // 0..8191 fragment entries
    if (idx < 6144) {
        int m  = idx / 2048;
        int r  = idx % 2048;
        int c  = r / 256;          // k-chunk 0..7
        int r2 = r % 256;
        int t  = r2 / 32;          // col tile 0..7
        int L  = r2 % 32;          // lane
        const float* W = (m == 0) ? Wq : ((m == 1) ? Wk : Wv);
        _Float16* dst = wsW + (size_t)m * 32768 + ((size_t)(c * 8 + t) * 32 + L) * 16;
        int col   = t * 16 + (L & 15);
        int kbase = c * 32 + ((L < 16) ? 0 : 8);
        v16h v;
#pragma unroll
        for (int h = 0; h < 16; ++h) v[h] = (_Float16)W[(size_t)(kbase + (h < 8 ? h : h + 8)) * 128 + col];
        *(v16h*)dst = v;
    } else {
        int e = idx - 6144;        // 0..2047
        int c = e / 512;           // k-chunk 0..3
        int r = e % 512;
        int t = r / 32;            // col tile 0..15
        int L = r % 32;
        _Float16* dst = wsW + (size_t)3 * 32768 + ((size_t)(c * 16 + t) * 32 + L) * 16;
        int col   = t * 16 + (L & 15);
        int kbase = c * 32 + ((L < 16) ? 0 : 8);
        v16h v;
#pragma unroll
        for (int h = 0; h < 16; ++h) v[h] = (_Float16)Wfc[(size_t)(kbase + (h < 8 ? h : h + 8)) * 256 + col];
        *(v16h*)dst = v;
    }
}

// ---------------------------------------------------------------------------
// Kernel 2: QKV projection. X[8192,256]f32 @ W[256,128] -> f16.
// One wave per 16x16 output tile (wave w = head w). 8 WMMA over K=256.
// A-frag 16x32: lane L -> row = L&15; halves: K=(L>>4)*8 + (h<8 ? h : 8+h).
// Q/K stored [b][h][s][dk]; V stored transposed [b][h][dk][s].
// ---------------------------------------------------------------------------
__global__ __launch_bounds__(256) void qkv_proj_kernel(
    const float* __restrict__ X, const _Float16* __restrict__ Wp,
    _Float16* __restrict__ dst, int v_transposed) {
    int lane = threadIdx.x & 31;
    int w    = threadIdx.x >> 5;     // head / col-tile 0..7
    int hi   = lane >> 4;            // half-wave id
    int ln   = lane & 15;
    int s0   = blockIdx.x * 16;      // global row base (bs index)

    v8f acc;
#pragma unroll
    for (int r = 0; r < 8; ++r) acc[r] = 0.f;

    const float* xrow = X + (size_t)(s0 + ln) * 256;
#pragma unroll
    for (int c = 0; c < 8; ++c) {
        int base = c * 32 + hi * 8;
        v4f f0 = *(const v4f*)(xrow + base);
        v4f f1 = *(const v4f*)(xrow + base + 4);
        v4f f2 = *(const v4f*)(xrow + base + 16);
        v4f f3 = *(const v4f*)(xrow + base + 20);
        v16h a;
#pragma unroll
        for (int j = 0; j < 4; ++j) {
            a[j]      = (_Float16)f0[j];
            a[4 + j]  = (_Float16)f1[j];
            a[8 + j]  = (_Float16)f2[j];
            a[12 + j] = (_Float16)f3[j];
        }
        v16h b = *(const v16h*)(Wp + ((size_t)(c * 8 + w) * 32 + lane) * 16);
        acc = wmma16(a, b, acc);
    }

    if (!v_transposed) {
        // [b][head][s][dk] ; D-tile: lane col = dk, rows = s
#pragma unroll
        for (int r = 0; r < 8; ++r) {
            int bs = s0 + hi * 8 + r;
            int b  = bs >> 11, s = bs & (S_LEN - 1);
            dst[((size_t)(b * NH + w) * S_LEN + s) * 16 + ln] = (_Float16)acc[r];
        }
    } else {
        // [b][head][dk][s] : per-lane contiguous 8 halves in s
        int b     = s0 >> 11;
        int sbase = (s0 & (S_LEN - 1)) + hi * 8;
        v8h o;
#pragma unroll
        for (int r = 0; r < 8; ++r) o[r] = (_Float16)acc[r];
        *(v8h*)(dst + ((size_t)(b * NH + w) * 16 + ln) * S_LEN + sbase) = o;
    }
}

// ---------------------------------------------------------------------------
// Kernel 3: flash attention. Block = (b, 16-query block); wave w = head.
// S^T = K * Q^T (d_k padded 16->32), online softmax in exp2 domain,
// ctx^T accumulated as V^T * P^T  (alpha rescale is per-lane scalar).
// ---------------------------------------------------------------------------
__global__ __launch_bounds__(256) void attn_kernel(
    const _Float16* __restrict__ Qh, const _Float16* __restrict__ Kh,
    const _Float16* __restrict__ Vt, const int* __restrict__ mask,
    _Float16* __restrict__ ctx) {
    const float SCL = 0.25f * 1.44269504088896340736f;  // 1/sqrt(16) * log2(e)
    int lane = threadIdx.x & 31;
    int w    = threadIdx.x >> 5;                 // head
    int hi   = lane >> 4, ln = lane & 15;
    int blk  = blockIdx.x;
    int b    = blk >> 7;                         // 128 q-blocks per batch
    int q0   = (blk & 127) * 16;

    // Q^T B-fragment: halves 0..7 hold K(=d)=hi*8..hi*8+7, halves 8..15 are zero pad.
    v16h qf;
    v8h qv = *(const v8h*)(Qh + ((size_t)(b * NH + w) * S_LEN + q0 + ln) * 16 + hi * 8);
#pragma unroll
    for (int h = 0; h < 8; ++h) { qf[h] = qv[h]; qf[8 + h] = (_Float16)0.f; }

    const _Float16* Kbase = Kh + (size_t)(b * NH + w) * S_LEN * 16;
    const _Float16* Vrow  = Vt + ((size_t)(b * NH + w) * 16 + ln) * S_LEN;  // ln = d
    const int*      mrow  = mask + ((size_t)b * S_LEN + (q0 + ln)) * S_LEN; // ln = q

    v8f acc;
#pragma unroll
    for (int r = 0; r < 8; ++r) acc[r] = 0.f;
    float m_run = -1e30f, l_run = 0.f;

    for (int kb = 0; kb < S_LEN; kb += 32) {
        // two 16x16 S^T tiles: rows = keys, cols = queries
        v8f st0, st1;
        {
            v8h kv0 = *(const v8h*)(Kbase + (size_t)(kb + ln) * 16 + hi * 8);
            v8h kv1 = *(const v8h*)(Kbase + (size_t)(kb + 16 + ln) * 16 + hi * 8);
            v16h a0, a1;
#pragma unroll
            for (int j = 0; j < 8; ++j) {
                a0[j] = kv0[j]; a0[8 + j] = (_Float16)0.f;   // K-dim 16..31 zero pad
                a1[j] = kv1[j]; a1[8 + j] = (_Float16)0.f;
            }
            v8f z;
#pragma unroll
            for (int r = 0; r < 8; ++r) z[r] = 0.f;
            st0 = wmma16(a0, qf, z);
            st1 = wmma16(a1, qf, z);
        }
        // mask + scale (log2 domain); lane holds keys k0..k0+7 of each tile
        float s0v[8], s1v[8];
        {
            int k0 = kb + hi * 8;
            v4i m0a = *(const v4i*)(mrow + k0);
            v4i m0b = *(const v4i*)(mrow + k0 + 4);
            v4i m1a = *(const v4i*)(mrow + k0 + 16);
            v4i m1b = *(const v4i*)(mrow + k0 + 20);
#pragma unroll
            for (int r = 0; r < 4; ++r) {
                s0v[r]     = m0a[r] ? -1e9f : st0[r] * SCL;
                s0v[4 + r] = m0b[r] ? -1e9f : st0[4 + r] * SCL;
                s1v[r]     = m1a[r] ? -1e9f : st1[r] * SCL;
                s1v[4 + r] = m1b[r] ? -1e9f : st1[4 + r] * SCL;
            }
        }
        // online softmax: lane pair (L, L^16) shares one query
        float mc = s0v[0];
#pragma unroll
        for (int r = 1; r < 8; ++r) mc = fmaxf(mc, s0v[r]);
#pragma unroll
        for (int r = 0; r < 8; ++r) mc = fmaxf(mc, s1v[r]);
        mc = fmaxf(mc, __shfl_xor(mc, 16, 32));
        float m_new = fmaxf(m_run, mc);
        float alpha = __builtin_amdgcn_exp2f(m_run - m_new);
        float p0[8], p1[8], sum = 0.f;
#pragma unroll
        for (int r = 0; r < 8; ++r) {
            p0[r] = __builtin_amdgcn_exp2f(s0v[r] - m_new);
            p1[r] = __builtin_amdgcn_exp2f(s1v[r] - m_new);
            sum += p0[r] + p1[r];
        }
        sum  += __shfl_xor(sum, 16, 32);
        l_run = l_run * alpha + sum;
        m_run = m_new;
#pragma unroll
        for (int r = 0; r < 8; ++r) acc[r] *= alpha;

        // P^T B-fragment: lanes<16 need keys 0..15 (own p0 + partner p0),
        // lanes>=16 need keys 16..31 (partner p1 + own p1): one xor-16 swap.
        v16h pf;
#pragma unroll
        for (int r = 0; r < 8; ++r) {
            float send = (lane < 16) ? p1[r] : p0[r];
            float recv = __shfl_xor(send, 16, 32);
            pf[r]     = (_Float16)((lane < 16) ? p0[r] : recv);
            pf[8 + r] = (_Float16)((lane < 16) ? recv  : p1[r]);
        }
        // V^T A-fragment: lane row = d, halves = keys (contiguous from Vt)
        v8h v0 = *(const v8h*)(Vrow + kb + hi * 16);
        v8h v1 = *(const v8h*)(Vrow + kb + hi * 16 + 8);
        v16h vf;
#pragma unroll
        for (int j = 0; j < 8; ++j) { vf[j] = v0[j]; vf[8 + j] = v1[j]; }

        acc = wmma16(vf, pf, acc);   // ctx^T += V^T * P^T
    }

    // ctx^T D-tile: lane col = query (= ln), rows = d -> contiguous b128 store
    float inv = 1.f / l_run;
    v8h o;
#pragma unroll
    for (int r = 0; r < 8; ++r) o[r] = (_Float16)(acc[r] * inv);
    *(v8h*)(ctx + ((size_t)(b * S_LEN) + q0 + ln) * 128 + w * 16 + hi * 8) = o;
}

// ---------------------------------------------------------------------------
// Kernel 4: FC (ctx[16,128] @ Wfc[128,256]) + LayerNorm per 256-wide row.
// Wave w owns col-tiles 2w, 2w+1 (8 WMMA); results staged in LDS f32,
// then wave w normalizes rows 2w, 2w+1 via shuffle reductions.
// ---------------------------------------------------------------------------
__global__ __launch_bounds__(256) void fc_ln_kernel(
    const _Float16* __restrict__ ctx, const _Float16* __restrict__ Wp,
    const float* __restrict__ gamma, const float* __restrict__ beta,
    float* __restrict__ out) {
    __shared__ float sm[16 * 260];
    int lane = threadIdx.x & 31;
    int w    = threadIdx.x >> 5;
    int hi   = lane >> 4, ln = lane & 15;
    int row0 = blockIdx.x * 16;
    int t0 = w * 2, t1 = w * 2 + 1;

    v8f acc0, acc1;
#pragma unroll
    for (int r = 0; r < 8; ++r) { acc0[r] = 0.f; acc1[r] = 0.f; }

    const _Float16* arow = ctx + (size_t)(row0 + ln) * 128;
#pragma unroll
    for (int c = 0; c < 4; ++c) {
        int base = c * 32 + hi * 8;
        v8h a0 = *(const v8h*)(arow + base);
        v8h a1 = *(const v8h*)(arow + base + 16);
        v16h a;
#pragma unroll
        for (int j = 0; j < 8; ++j) { a[j] = a0[j]; a[8 + j] = a1[j]; }
        v16h b0 = *(const v16h*)(Wp + ((size_t)(c * 16 + t0) * 32 + lane) * 16);
        v16h b1 = *(const v16h*)(Wp + ((size_t)(c * 16 + t1) * 32 + lane) * 16);
        acc0 = wmma16(a, b0, acc0);
        acc1 = wmma16(a, b1, acc1);
    }
#pragma unroll
    for (int r = 0; r < 8; ++r) {
        int srow = hi * 8 + r;
        sm[srow * 260 + t0 * 16 + ln] = acc0[r];
        sm[srow * 260 + t1 * 16 + ln] = acc1[r];
    }
    __syncthreads();

#pragma unroll
    for (int i = 0; i < 2; ++i) {
        int row = w * 2 + i;
        float v[8], sum = 0.f, sq = 0.f;
#pragma unroll
        for (int j = 0; j < 8; ++j) {
            v[j] = sm[row * 260 + lane + 32 * j];
            sum += v[j];
            sq  += v[j] * v[j];
        }
#pragma unroll
        for (int off = 16; off >= 1; off >>= 1) {
            sum += __shfl_xor(sum, off, 32);
            sq  += __shfl_xor(sq,  off, 32);
        }
        float mean = sum * (1.f / 256.f);
        float var  = sq  * (1.f / 256.f) - mean * mean;
        float rs   = rsqrtf(var + 1e-5f);
        float* orow = out + (size_t)(row0 + row) * 256;
#pragma unroll
        for (int j = 0; j < 8; ++j) {
            int c = lane + 32 * j;
            orow[c] = (v[j] - mean) * rs * gamma[c] + beta[c];
        }
    }
}

// ---------------------------------------------------------------------------
extern "C" void kernel_launch(void* const* d_in, const int* in_sizes, int n_in,
                              void* d_out, int out_size, void* d_ws, size_t ws_size,
                              hipStream_t stream) {
    const float* inQ  = (const float*)d_in[0];
    const float* inK  = (const float*)d_in[1];
    const float* inV  = (const float*)d_in[2];
    const int*   mask = (const int*)d_in[3];
    const float* Wq   = (const float*)d_in[4];
    const float* Wk   = (const float*)d_in[5];
    const float* Wv   = (const float*)d_in[6];
    const float* Wfc  = (const float*)d_in[7];
    const float* gam  = (const float*)d_in[8];
    const float* bet  = (const float*)d_in[9];
    float* out = (float*)d_out;

    _Float16* ws   = (_Float16*)d_ws;
    _Float16* WqP  = ws;                 // 32768 halves each
    _Float16* WkP  = ws + 32768;
    _Float16* WvP  = ws + 65536;
    _Float16* WfcP = ws + 98304;
    _Float16* Qh   = ws + 131072;        // [4][8][2048][16] f16
    _Float16* Kh   = Qh + 1048576;
    _Float16* Vt   = Kh + 1048576;       // [4][8][16][2048] f16
    _Float16* ctx  = Vt + 1048576;       // [4][2048][128]   f16

    pack_weights_kernel<<<32, 256, 0, stream>>>(Wq, Wk, Wv, Wfc, ws);
    qkv_proj_kernel<<<512, 256, 0, stream>>>(inQ, WqP, Qh, 0);
    qkv_proj_kernel<<<512, 256, 0, stream>>>(inK, WkP, Kh, 0);
    qkv_proj_kernel<<<512, 256, 0, stream>>>(inV, WvP, Vt, 1);
    attn_kernel<<<512, 256, 0, stream>>>(Qh, Kh, Vt, mask, ctx);
    fc_ln_kernel<<<512, 256, 0, stream>>>(ctx, WfcP, gam, bet, out);
}
